// Kernel_Layer_55654186221945
// MI455X (gfx1250) — matrix operand order repaired and checked by reading, not compiled, not run on hardware
//
#include <hip/hip_runtime.h>
#include <stdint.h>

// ---------------------------------------------------------------------------
// bmi,mio->bmo grouped GEMM, fp32 in/out, bf16x3 split-precision WMMA core.
// BATCH=8192, M=16, K(D_IN)=256, N(D_OUT)=256.
//
// Roofline: 17.2 GFLOP vs 260 MiB HBM traffic -> 11.7us memory floor at
// 23.3 TB/s.  Native f32 WMMA (16x16x4) would be ~4-5x compute-bound; the
// bf16 split (hi*hi + lo*hi + hi*lo at 16x16x32 rate) lands at ~2x the
// memory floor with ~2^-16 relative error.  Weights (4 MiB) are L2-resident
// and pre-split once per launch into WMMA B-fragment order.
// ---------------------------------------------------------------------------

typedef __attribute__((ext_vector_type(16))) __bf16 v16bf;
typedef __attribute__((ext_vector_type(8)))  float  v8f;

#define BATCH 8192
#define MDIM  16
#define DIN   256
#define DOUT  256

// Pre-packed weight fragments, bf16 hi/lo, in WMMA 16x16x32 B-matrix lane
// order.  Index: ((m*16 + ct)*8 + s)*64 + i*32 + lane, each a uint4
// (8 bf16 = elements i*8 .. i*8+7 of the v16bf fragment for that lane).
__device__ uint4 g_whi[16 * 16 * 8 * 64];
__device__ uint4 g_wlo[16 * 16 * 8 * 64];

union BF16Frag {
    v16bf          v;
    unsigned       d[8];
    uint4          q[2];
    unsigned short u[16];
};

static __device__ __forceinline__ unsigned short bf_rne(float f) {
    unsigned x = __float_as_uint(f);
    x += 0x7FFFu + ((x >> 16) & 1u);
    return (unsigned short)(x >> 16);
}
static __device__ __forceinline__ float bf_to_f(unsigned short h) {
    return __uint_as_float(((unsigned)h) << 16);
}

// Cheap truncation split of a float pair into packed-bf16 hi and lo dwords.
// hi = f with low 16 mantissa bits dropped (exact as bf16), lo = trunc-bf16
// of the residual.  One v_perm_b32 packs two high halves in a single op.
static __device__ __forceinline__ void split_pair(float f0, float f1,
                                                  unsigned& hi_pk,
                                                  unsigned& lo_pk) {
    unsigned u0 = __float_as_uint(f0);
    unsigned u1 = __float_as_uint(f1);
    hi_pk = __builtin_amdgcn_perm(u1, u0, 0x07060302u);  // {u1.hi16, u0.hi16}
    float h0 = __uint_as_float(u0 & 0xffff0000u);
    float h1 = __uint_as_float(u1 & 0xffff0000u);
    float l0 = f0 - h0;
    float l1 = f1 - h1;
    lo_pk = __builtin_amdgcn_perm(__float_as_uint(l1), __float_as_uint(l0),
                                  0x07060302u);
}

// Build A hi/lo fragments (16 elements in WMMA element order) from four
// float4 loads.
static __device__ __forceinline__ void build_a(const float* p, BF16Frag& h,
                                               BF16Frag& l) {
    const float4* q = (const float4*)p;
    float4 f0 = q[0], f1 = q[1], f2 = q[4], f3 = q[5];
    split_pair(f0.x, f0.y, h.d[0], l.d[0]);
    split_pair(f0.z, f0.w, h.d[1], l.d[1]);
    split_pair(f1.x, f1.y, h.d[2], l.d[2]);
    split_pair(f1.z, f1.w, h.d[3], l.d[3]);
    split_pair(f2.x, f2.y, h.d[4], l.d[4]);
    split_pair(f2.z, f2.w, h.d[5], l.d[5]);
    split_pair(f3.x, f3.y, h.d[6], l.d[6]);
    split_pair(f3.z, f3.w, h.d[7], l.d[7]);
}

static __device__ __forceinline__ v8f wmma_bf16(v16bf a, v16bf b, v8f c) {
    return __builtin_amdgcn_wmma_f32_16x16x32_bf16(false, a, false, b,
                                                   (short)0, c, false, false);
}

// ---------------------------------------------------------------------------
// Pre-pass: split fp32 weights (M, DIN, DOUT) into bf16 hi/lo fragments in
// B-matrix VGPR order (lane n<16: column o=ct*16+n, K=s*32+0..7,16..23; lanes
// 16-31: K=s*32+8..15,24..31).  One thread per (m, ct, s, lane) = 65536 threads.
// Uses RNE here since this runs once over 4 MiB (cost is negligible).
// ---------------------------------------------------------------------------
__global__ void __launch_bounds__(256) pack_weights(const float* __restrict__ w) {
    int tid  = blockIdx.x * blockDim.x + threadIdx.x;
    int lane = tid & 31;
    int s    = (tid >> 5) & 7;
    int ct   = (tid >> 8) & 15;
    int m    = tid >> 12;

    int k0 = s * 32 + ((lane >> 4) << 3);
    int o  = ct * 16 + (lane & 15);

    const float* p = w + (size_t)m * DIN * DOUT + (size_t)k0 * DOUT + o;

    BF16Frag hi, lo;
#pragma unroll
    for (int j = 0; j < 16; ++j) {
        float f = p[(size_t)(j + ((j >> 3) << 3)) * DOUT];
        unsigned short h = bf_rne(f);
        hi.u[j] = h;
        lo.u[j] = bf_rne(f - bf_to_f(h));
    }

    int base = ((m * 16 + ct) * 8 + s) * 64;
    g_whi[base + lane]      = hi.q[0];
    g_whi[base + 32 + lane] = hi.q[1];
    g_wlo[base + lane]      = lo.q[0];
    g_wlo[base + 32 + lane] = lo.q[1];
}

// ---------------------------------------------------------------------------
// Main kernel: each wave computes a 32x128 output block (2 row-tiles x 8
// col-tiles, fixed m).  48 WMMAs per K-step of 32 against ~170 split/addr
// VALU ops -> matrix-pipe bound.  8192 waves = 1024 blocks x 256 threads.
// ---------------------------------------------------------------------------
__global__ void __launch_bounds__(256) grouped_gemm(const float* __restrict__ x,
                                                    float* __restrict__ out) {
    int wave = (int)((blockIdx.x * blockDim.x + threadIdx.x) >> 5); // 0..8191
    int lane = threadIdx.x & 31;

    int ch = wave & 1;          // column half: 8 tiles => 128 output columns
    int m  = (wave >> 1) & 15;  // which weight matrix
    int rp = wave >> 5;         // 0..255, 32 batch rows each

    int lr = lane & 15;
    int lh = lane >> 4;

    // A rows for this lane; 16-bit A layout: lanes 0-15 hold K 0..7 & 16..23,
    // lanes 16-31 hold K 8..15 & 24..31 -> base offset lh*8, second run +16.
    int b0 = rp * 32 + lr;
    const float* xr0 = x + ((size_t)b0 * MDIM + m) * DIN + lh * 8;
    const float* xr1 = xr0 + (size_t)16 * MDIM * DIN;   // row tile 1 (+16 rows)

    v8f c[2][8];
#pragma unroll
    for (int rt = 0; rt < 2; ++rt)
#pragma unroll
        for (int t = 0; t < 8; ++t)
            c[rt][t] = (v8f)0.0f;

    int wbase = ((m * 16 + ch * 8) * 8) * 64;   // fragment base for ct = ch*8

#pragma unroll 1
    for (int s = 0; s < 8; ++s) {               // K-steps of 32
        BF16Frag a0h, a0l, a1h, a1l;
        build_a(xr0 + s * 32, a0h, a0l);
        build_a(xr1 + s * 32, a1h, a1l);

#pragma unroll
        for (int t = 0; t < 8; ++t) {
            int fb = wbase + (t * 8 + s) * 64;
            BF16Frag bh, bl;
            bh.q[0] = g_whi[fb + lane];
            bh.q[1] = g_whi[fb + 32 + lane];
            bl.q[0] = g_wlo[fb + lane];
            bl.q[1] = g_wlo[fb + 32 + lane];

            // Alternate accumulators so back-to-back WMMAs are independent.
            c[0][t] = wmma_bf16(a0h.v, bh.v, c[0][t]);   // hi*hi
            c[1][t] = wmma_bf16(a1h.v, bh.v, c[1][t]);
            c[0][t] = wmma_bf16(a0l.v, bh.v, c[0][t]);   // lo*hi
            c[1][t] = wmma_bf16(a1l.v, bh.v, c[1][t]);
            c[0][t] = wmma_bf16(a0h.v, bl.v, c[0][t]);   // hi*lo
            c[1][t] = wmma_bf16(a1h.v, bl.v, c[1][t]);
        }
    }

    // Store.  C/D layout: VGPR v holds (row = v + 8*lh, col = lr) of a tile.
    // Single base pointer + constant offsets (fold into store immediates).
    float* ob = out + (((size_t)(rp * 32 + lh * 8) * MDIM + m) * DOUT)
                    + ch * 128 + lr;
#pragma unroll
    for (int rt = 0; rt < 2; ++rt)
#pragma unroll
        for (int t = 0; t < 8; ++t)
#pragma unroll
            for (int v = 0; v < 8; ++v)
                ob[(size_t)(rt * 16 + v) * (MDIM * DOUT) + t * 16] = c[rt][t][v];
}

// ---------------------------------------------------------------------------
extern "C" void kernel_launch(void* const* d_in, const int* in_sizes, int n_in,
                              void* d_out, int out_size, void* d_ws, size_t ws_size,
                              hipStream_t stream) {
    (void)in_sizes; (void)n_in; (void)out_size; (void)d_ws; (void)ws_size;
    const float* x = (const float*)d_in[0];   // (8192, 16, 256) fp32
    const float* w = (const float*)d_in[1];   // (16, 256, 256) fp32
    float* out     = (float*)d_out;           // (8192, 16, 256) fp32

    pack_weights<<<256, 256, 0, stream>>>(w);
    grouped_gemm<<<1024, 256, 0, stream>>>(x, out);
}
